// BondCenteredTensorMomentDescriptor_3367254360422
// MI455X (gfx1250) — matrix operand order repaired and checked by reading, not compiled, not run on hardware
//
#include <hip/hip_runtime.h>
#include <math.h>

#define NE   100000
#define EPW  16          // edges per wave
#define WPB  4           // waves per block
#define EPB  (EPW*WPB)

typedef __attribute__((ext_vector_type(16))) _Float16 v16h;
typedef __attribute__((ext_vector_type(8)))  _Float16 v8h;
typedef __attribute__((ext_vector_type(8)))  float    v8f;

// ---------------- compile-time Clebsch-Gordan tables ----------------
struct CD { double re, im; };
constexpr CD cmul(CD a, CD b){ return { a.re*b.re - a.im*b.im, a.re*b.im + a.im*b.re }; }
constexpr CD cscale(CD a, double s){ return { a.re*s, a.im*s }; }
constexpr CD cadd(CD a, CD b){ return { a.re+b.re, a.im+b.im }; }
constexpr CD cconj(CD a){ return { a.re, -a.im }; }
constexpr double factd(int n){ double r=1.0; for(int i=2;i<=n;++i) r*=(double)i; return r; }
constexpr double csqrt_(double x){ double g = x < 1.0 ? 1.0 : x; for(int i=0;i<64;++i) g = 0.5*(g + x/g); return g; }
constexpr int iabs_(int x){ return x<0?-x:x; }

constexpr double cg_complex(int j1,int m1,int j2,int m2,int j3,int m3){
  if (m1+m2 != m3) return 0.0;
  if (j3 < iabs_(j1-j2) || j3 > j1+j2) return 0.0;
  if (iabs_(m1)>j1 || iabs_(m2)>j2 || iabs_(m3)>j3) return 0.0;
  double pref = csqrt_((2.0*j3+1.0)*factd(j3+j1-j2)*factd(j3-j1+j2)*factd(j1+j2-j3)/factd(j1+j2+j3+1));
  pref *= csqrt_(factd(j3+m3)*factd(j3-m3)*factd(j1-m1)*factd(j1+m1)*factd(j2-m2)*factd(j2+m2));
  double s = 0.0;
  for (int k=0;k<=j1+j2-j3;++k){
    int d1=j1+j2-j3-k, d2=j1-m1-k, d3=j2+m2-k, d4=j3-j2+m1+k, d5=j3-j1-m2+k;
    if (d1<0||d2<0||d3<0||d4<0||d5<0) continue;
    double t = 1.0/(factd(k)*factd(d1)*factd(d2)*factd(d3)*factd(d4)*factd(d5));
    s += (k&1) ? -t : t;
  }
  return pref*s;
}

struct UM { CD u[9][9]; };
constexpr UM makeU(int l){
  UM U{};
  const double is2 = 1.0/csqrt_(2.0);
  for (int m=-l;m<=l;++m){
    int i = m+l;
    if (m>0){
      U.u[i][m+l]  = { (m&1)? -is2 : is2, 0.0 };
      U.u[i][-m+l] = { is2, 0.0 };
    } else if (m==0){
      U.u[i][l] = { 1.0, 0.0 };
    } else {
      U.u[i][m+l]  = { 0.0, is2 };
      U.u[i][-m+l] = { 0.0, ((-m)&1)? is2 : -is2 };
    }
  }
  return U;
}

template<int L1,int L2,int L3> struct CGHold { float c[(2*L1+1)*(2*L2+1)*(2*L3+1)]; };

template<int L1,int L2,int L3>
constexpr CGHold<L1,L2,L3> make_cg(){
  constexpr int D1=2*L1+1, D2=2*L2+1, D3=2*L3+1;
  double base[D1][D2][D3] = {};
  for (int m1=-L1;m1<=L1;++m1)
    for (int m2=-L2;m2<=L2;++m2){
      int m3 = m1+m2;
      if (iabs_(m3) <= L3) base[m1+L1][m2+L2][m3+L3] = cg_complex(L1,m1,L2,m2,L3,m3);
    }
  UM U1 = makeU(L1), U2 = makeU(L2), U3 = makeU(L3);
  CD T1[D1][D2][D3] = {};
  for (int a=0;a<D1;++a) for (int x=0;x<D1;++x){
    CD w = cconj(U1.u[a][x]);
    if (w.re==0.0 && w.im==0.0) continue;
    for (int y=0;y<D2;++y) for (int z=0;z<D3;++z)
      T1[a][y][z] = cadd(T1[a][y][z], cscale(w, base[x][y][z]));
  }
  CD T2[D1][D2][D3] = {};
  for (int b=0;b<D2;++b) for (int y=0;y<D2;++y){
    CD w = cconj(U2.u[b][y]);
    if (w.re==0.0 && w.im==0.0) continue;
    for (int a=0;a<D1;++a) for (int z=0;z<D3;++z)
      T2[a][b][z] = cadd(T2[a][b][z], cmul(w, T1[a][y][z]));
  }
  CGHold<L1,L2,L3> H{};
  for (int a=0;a<D1;++a) for (int b=0;b<D2;++b) for (int c=0;c<D3;++c){
    CD acc{0.0,0.0};
    for (int z=0;z<D3;++z) acc = cadd(acc, cmul(U3.u[c][z], T2[a][b][z]));
    double v = ((L1+L2+L3)&1) ? acc.im : acc.re;
    if (v < 1e-10 && v > -1e-10) v = 0.0;
    H.c[(a*D2+b)*D3+c] = (float)v;
  }
  return H;
}

template<int L1,int L2,int L3>
struct CGC { static constexpr CGHold<L1,L2,L3> v = make_cg<L1,L2,L3>(); };

// ---------------- per-wave LDS layout (bytes) ----------------
#define OFF_ACT 0           // 144*40 f16 = 11520
#define OFF_Y   11520       // 144*33 f32 = 19008
#define OFF_RAD 30528       // 16*40 f16  = 1280
#define OFF_T   31808       // 16*33 f32  = 2112
#define OFF_SH  33920       // 16*16 f32  = 1024
#define WS_BYTES 34944

__device__ inline v8f wmma16(v16h a, v16h b, v8f c){
  return __builtin_amdgcn_wmma_f32_16x16x32_f16(false, a, false, b, (short)0, c, false, false);
}
__device__ inline v16h cat8(v8h lo, v8h hi){
  return __builtin_shufflevector(lo, hi, 0,1,2,3,4,5,6,7,8,9,10,11,12,13,14,15);
}
__device__ inline v16h lds_afrag(const _Float16* row, int kc){
  v8h lo = *(const v8h*)(row + kc);
  v8h hi = *(const v8h*)(row + kc + 16);
  return cat8(lo, hi);
}
__device__ inline float wsum32(float v){
  #pragma unroll
  for (int m=1;m<32;m<<=1) v += __shfl_xor(v, m, 32);
  return v;
}

#define TP_PATH(L1,L2,L3,P) do { \
  constexpr int PAR = ((L1)+(L2)+(L3)) & 1; \
  const float tw = tpr[P]; \
  _Pragma("unroll") \
  for (int a=0; a<2*(L1)+1; ++a) { \
    _Pragma("unroll") \
    for (int b=0; b<2*(L2)+1; ++b) { \
      const float pab = bexp[(L1)*(L1)+a] * yv[(L2)*(L2)+b] * tw; \
      _Pragma("unroll") \
      for (int c=0; c<2*(L3)+1; ++c) { \
        const float cv = CGC<L1,L2,L3>::v.c[(a*(2*(L2)+1)+b)*(2*(L3)+1)+c]; \
        if (cv != 0.0f) acc[PAR*25 + (L3)*(L3) + c] += cv * pab; \
      } } } \
} while(0)

__global__ __launch_bounds__(WPB*32, 1)
void bctmd_fused(const float* __restrict__ desc,
                 const int*   __restrict__ nbr,
                 const float* __restrict__ disp,
                 const float* __restrict__ W1, const float* __restrict__ b1,
                 const float* __restrict__ gam, const float* __restrict__ bet,
                 const float* __restrict__ W2, const float* __restrict__ b2,
                 const float* __restrict__ Wb, const float* __restrict__ bb,
                 const float* __restrict__ tpw,
                 float* __restrict__ out)
{
  extern __shared__ char smem[];
  const int lane = threadIdx.x & 31;
  const int wid  = threadIdx.x >> 5;
  const int e0   = (blockIdx.x * WPB + wid) * EPW;
  if (e0 >= NE) return;                       // uniform per wave

  char* wbase = smem + wid * WS_BYTES;
  _Float16* Sact = (_Float16*)(wbase + OFF_ACT);
  float*    Sy   = (float*)   (wbase + OFF_Y);
  _Float16* Srad = (_Float16*)(wbase + OFF_RAD);
  float*    St   = (float*)   (wbase + OFF_T);
  float*    Ssh  = (float*)   (wbase + OFF_SH);

  const int rowm = lane & 15;
  const int kh   = lane >> 4;
  const int kc   = kh * 8;

  { // pull edge metadata toward the caches
    int pe = e0 + (lane % EPW); if (pe >= NE) pe = NE-1;
    __builtin_prefetch(&nbr[2*pe], 0, 1);
    __builtin_prefetch(&disp[3*pe], 0, 1);
  }

  // per-lane constants
  const float b1lo = b1[rowm], b1hi = b1[rowm+16];
  const float b2lo = b2[rowm], b2hi = b2[rowm+16];
  const float bbl  = bb[lane];
  const float g0 = gam[lane], g1 = gam[32+lane], g2 = gam[64+lane];
  const float betl = bet[lane];
  float tpr[19];
  #pragma unroll
  for (int p=0;p<19;++p) tpr[p] = tpw[p*32 + lane];

  // B fragments: lane = column N, K split 8h+s / 16+8h+s (matches A)
  auto loadB = [&](const float* W, int nt)->v16h {
    const int n  = rowm + 16*nt;
    const int kb = kh * 8;
    v16h r;
    #pragma unroll
    for (int k=0;k<16;++k) r[k] = (_Float16)W[(kb + (k<8 ? k : k+8))*32 + n];
    return r;
  };
  const v16h B1a = loadB(W1,0), B1b = loadB(W1,1);
  const v16h B2a = loadB(W2,0), B2b = loadB(W2,1);
  const v16h Bba = loadB(Wb,0), Bbb = loadB(Wb,1);
  const v8f vzero = {};

  // ---- Phase 1: gather desc[i]+desc[j], dense1 (WMMA), y0 -> LDS ----
  #pragma unroll 3
  for (int mt=0; mt<9; ++mt){
    const int arow = mt*16 + rowm;            // A-fragment row = M
    const int ael  = arow / 9;
    const int am   = arow - ael*9;
    int e = e0 + ael; if (e >= NE) e = NE-1;
    const int ia = nbr[2*e], ja = nbr[2*e+1];
    const float* pi = desc + (size_t)ia*288 + am*32;
    const float* pj = desc + (size_t)ja*288 + am*32;
    float va[16], vb[16];
    *(float4*)&va[0]  = *(const float4*)(pi + kc);
    *(float4*)&va[4]  = *(const float4*)(pi + kc + 4);
    *(float4*)&va[8]  = *(const float4*)(pi + kc + 16);
    *(float4*)&va[12] = *(const float4*)(pi + kc + 20);
    *(float4*)&vb[0]  = *(const float4*)(pj + kc);
    *(float4*)&vb[4]  = *(const float4*)(pj + kc + 4);
    *(float4*)&vb[8]  = *(const float4*)(pj + kc + 16);
    *(float4*)&vb[12] = *(const float4*)(pj + kc + 20);
    v16h A;
    #pragma unroll
    for (int q=0;q<16;++q) A[q] = (_Float16)(va[q] + vb[q]);
    v8f c0 = wmma16(A, B1a, vzero);
    v8f c1 = wmma16(A, B1b, vzero);
    #pragma unroll
    for (int r=0;r<8;++r){
      const int rr  = mt*16 + r + 8*kh;       // D row
      const int rel = rr / 9;
      const int rm  = rr - rel*9;
      const float ad = (rm==0) ? 1.0f : 0.0f;
      Sy[rr*33 + rowm]      = c0[r] + ad*b1lo;
      Sy[rr*33 + rowm + 16] = c1[r] + ad*b1hi;
    }
  }
  asm volatile("" ::: "memory");

  // ---- Phase 2: layernorm + mish gate (lane = feature), act -> LDS f16 ----
  #pragma unroll 1
  for (int el=0; el<EPW; ++el){
    float y0[9];
    #pragma unroll
    for (int m=0;m<9;++m) y0[m] = Sy[(el*9+m)*33 + lane];
    const float s   = y0[0];
    const float mu  = wsum32(s) * (1.0f/32.0f);
    const float dc  = s - mu;
    const float var = wsum32(dc*dc) * (1.0f/32.0f);
    const float sn  = dc * rsqrtf(var + 1e-6f) * g0 + betl;
    const float q1  = y0[1]*y0[1] + y0[2]*y0[2] + y0[3]*y0[3];
    const float ir1 = rsqrtf(wsum32(q1)*(1.0f/96.0f) + 1e-6f);
    const float q2  = y0[4]*y0[4]+y0[5]*y0[5]+y0[6]*y0[6]+y0[7]*y0[7]+y0[8]*y0[8];
    const float ir2 = rsqrtf(wsum32(q2)*(1.0f/160.0f) + 1e-6f);
    const float sp  = (sn > 15.0f) ? sn : log1pf(expf(sn));
    const float th  = tanhf(sp);
    const float sg  = 1.0f/(1.0f + expf(-sn));
    const float dgt = th + sn*(1.0f - th*th)*sg;
    _Float16* ar = Sact + el*9*40;
    ar[lane] = (_Float16)(sn*th);
    #pragma unroll
    for (int m=1;m<4;++m) ar[m*40+lane] = (_Float16)(y0[m]*ir1*g1*dgt);
    #pragma unroll
    for (int m=4;m<9;++m) ar[m*40+lane] = (_Float16)(y0[m]*ir2*g2*dgt);
  }
  asm volatile("" ::: "memory");

  // ---- Phase 3: dense2 (WMMA, C = y0 + bias residual), y2 -> LDS ----
  #pragma unroll 3
  for (int mt=0; mt<9; ++mt){
    const _Float16* ap = Sact + (mt*16 + rowm)*40;
    v16h A = lds_afrag(ap, kc);
    v8f C0, C1;
    #pragma unroll
    for (int r=0;r<8;++r){
      const int rr  = mt*16 + r + 8*kh;
      const int rel = rr/9; const int rm = rr - rel*9;
      const float ad = (rm==0) ? 1.0f : 0.0f;
      C0[r] = Sy[rr*33 + rowm]      + ad*b2lo;
      C1[r] = Sy[rr*33 + rowm + 16] + ad*b2hi;
    }
    v8f d0 = wmma16(A, B2a, C0);
    v8f d1 = wmma16(A, B2b, C1);
    #pragma unroll
    for (int r=0;r<8;++r){
      const int rr = mt*16 + r + 8*kh;
      Sy[rr*33 + rowm]      = d0[r];
      Sy[rr*33 + rowm + 16] = d1[r];
    }
  }
  asm volatile("" ::: "memory");

  // ---- Phase 4: radial basis rows (lane = k), sph harm, rad@Wb via WMMA ----
  const float kctr = (float)lane * (5.0f/31.0f);
  const float kgam = 0.5f * (32.0f/5.0f) * (32.0f/5.0f);
  #pragma unroll 1
  for (int el=0; el<EPW; ++el){
    int e = e0 + el; if (e >= NE) e = NE-1;
    const float dx = disp[3*e], dy = disp[3*e+1], dz = disp[3*e+2];
    const float r  = sqrtf(dx*dx + dy*dy + dz*dz);
    const float dr = r - kctr;
    float cut = 0.0f;
    if (r < 5.0f) cut = 0.5f*(cosf(3.14159265358979323846f * r * 0.2f) + 1.0f);
    Srad[el*40 + lane] = (_Float16)(expf(-kgam*dr*dr) * cut);
    const float ri = 1.0f / fmaxf(r, 1e-12f);
    const float ux = dx*ri, uy = dy*ri, uz = dz*ri;
    if (lane < 9){
      const float s3 = 1.7320508075688772f;
      float sv;
      switch(lane){
        case 0: sv = 1.0f; break;
        case 1: sv = uy; break;
        case 2: sv = uz; break;
        case 3: sv = ux; break;
        case 4: sv = s3*ux*uy; break;
        case 5: sv = s3*uy*uz; break;
        case 6: sv = 0.5f*(3.0f*uz*uz - 1.0f); break;
        case 7: sv = s3*ux*uz; break;
        default: sv = 0.5f*s3*(ux*ux - uy*uy); break;
      }
      Ssh[el*16 + lane] = sv;
    }
  }
  asm volatile("" ::: "memory");
  {
    const _Float16* rp = Srad + rowm*40;
    v16h Ar = lds_afrag(rp, kc);
    v8f t0 = wmma16(Ar, Bba, vzero);
    v8f t1 = wmma16(Ar, Bbb, vzero);
    #pragma unroll
    for (int r=0;r<8;++r){
      const int er = r + 8*kh;
      St[er*33 + rowm]      = t0[r];
      St[er*33 + rowm + 16] = t1[r];
    }
  }
  asm volatile("" ::: "memory");

  // ---- Phase 5: tensor product (compile-time CG, fully unrolled) + store ----
  #pragma unroll 1
  for (int el=0; el<EPW; ++el){
    if (e0 + el >= NE) break;                 // uniform
    const float tq = St[el*33 + lane];
    float bexp[9];
    bexp[0] = tq + bbl;                       // sh0 == 1, bias on l=0 row
    #pragma unroll
    for (int a=1;a<9;++a) bexp[a] = Ssh[el*16 + a] * tq;
    float yv[9];
    #pragma unroll
    for (int b=0;b<9;++b) yv[b] = Sy[(el*9+b)*33 + lane];
    float acc[50];
    #pragma unroll
    for (int o=0;o<50;++o) acc[o] = 0.0f;

    TP_PATH(0,0,0,0);  TP_PATH(0,1,1,1);  TP_PATH(0,2,2,2);
    TP_PATH(1,0,1,3);  TP_PATH(1,1,0,4);  TP_PATH(1,1,1,5);  TP_PATH(1,1,2,6);
    TP_PATH(1,2,1,7);  TP_PATH(1,2,2,8);  TP_PATH(1,2,3,9);
    TP_PATH(2,0,2,10); TP_PATH(2,1,1,11); TP_PATH(2,1,2,12); TP_PATH(2,1,3,13);
    TP_PATH(2,2,0,14); TP_PATH(2,2,1,15); TP_PATH(2,2,2,16); TP_PATH(2,2,3,17); TP_PATH(2,2,4,18);

    float* op = out + (size_t)(e0+el)*1600 + lane;   // (E,2,25,32): o*32+lane
    #pragma unroll
    for (int o=0;o<50;++o) op[o*32] = acc[o];
  }
}

extern "C" void kernel_launch(void* const* d_in, const int* in_sizes, int n_in,
                              void* d_out, int out_size, void* d_ws, size_t ws_size,
                              hipStream_t stream)
{
  (void)in_sizes; (void)n_in; (void)out_size; (void)d_ws; (void)ws_size;
  const float* desc = (const float*)d_in[0];
  const int*   nbr  = (const int*)  d_in[1];
  const float* disp = (const float*)d_in[2];
  const float* W1   = (const float*)d_in[3];
  const float* b1   = (const float*)d_in[4];
  const float* gam  = (const float*)d_in[5];
  const float* bet  = (const float*)d_in[6];
  const float* W2   = (const float*)d_in[7];
  const float* b2   = (const float*)d_in[8];
  const float* Wb   = (const float*)d_in[9];
  const float* bb   = (const float*)d_in[10];
  const float* tpw  = (const float*)d_in[11];
  float* out = (float*)d_out;

  const int blocks = (NE + EPB - 1) / EPB;
  const size_t shmem = (size_t)WPB * WS_BYTES;
  bctmd_fused<<<blocks, WPB*32, shmem, stream>>>(desc, nbr, disp, W1, b1, gam, bet,
                                                 W2, b2, Wb, bb, tpw, out);
}
